// QuantizedLinearZSE_59373627900306
// MI455X (gfx1250) — compile-verified, matrix operand order checked by reading, not run on hardware
//
#include <hip/hip_runtime.h>

typedef __attribute__((ext_vector_type(16))) _Float16 v16h;
typedef __attribute__((ext_vector_type(8)))  _Float16 v8h;
typedef __attribute__((ext_vector_type(2)))  _Float16 h2;
typedef __attribute__((ext_vector_type(8)))  float    v8f;

#define M_DIM   8192
#define K_DIM   4096
#define N_DIM   11008
#define GROUPS  32            // K / 128
#define PK      (K_DIM / 2)   // int32 elements per packed row (one byte each)

#define TM_TILE 128
#define TN_TILE 128
#define BK      32
#define LDT     40            // padded LDS row stride in halves (80B, 16B aligned)
#define NT      (K_DIM / BK)  // 128 k-tiles

union V8u { v8h v; h2 h[4]; };

#define CVT_PK(a, b) __builtin_bit_cast(h2, __builtin_amdgcn_cvt_pkrtz((a), (b)))

__global__ __launch_bounds__(256)
void QuantizedLinearZSE_59373627900306_kernel(const float* __restrict__ x,
                                              const int*   __restrict__ packed,
                                              const float* __restrict__ scales,
                                              float*       __restrict__ out)
{
    __shared__ __align__(16) _Float16 As[2][TM_TILE * LDT];
    __shared__ __align__(16) _Float16 Bs[2][TN_TILE * LDT];

    const int tid = threadIdx.x;
    const int bm  = blockIdx.y * TM_TILE;
    const int bn  = blockIdx.x * TN_TILE;

    // staging: 2 threads per tile row, each covers 16 of the 32 k-columns
    const int r  = tid >> 1;
    const int ch = tid & 1;

    // wave layout: 4 waves along M (32 rows each), 2 waves along N (64 cols each)
    const int wid   = tid >> 5;
    const int lane  = tid & 31;
    const int wm    = (wid & 3) * 32;
    const int wn    = (wid >> 2) * 64;
    const int lhalf = lane >> 4;   // selects K sub-chunk per ISA A/B layout
    const int lrow  = lane & 15;   // row (A) / col (B) within 16

    v8f acc[2][4];
#pragma unroll
    for (int tm = 0; tm < 2; ++tm)
#pragma unroll
        for (int tn = 0; tn < 4; ++tn)
            acc[tm][tn] = (v8f){0.f, 0.f, 0.f, 0.f, 0.f, 0.f, 0.f, 0.f};

    const float* xrow = x      + (size_t)(bm + r) * K_DIM + ch * 16;
    const int*   prow = packed + (size_t)(bn + r) * PK    + ch * 8;
    const float* srow = scales + (size_t)(bn + r) * GROUPS;

    float4 xa0, xa1, xa2, xa3;
    int4   w0,  w1;
    h2     sc2, off2;

    auto gload = [&](int kt) {
        const float4* xp = reinterpret_cast<const float4*>(xrow + kt * BK);
        xa0 = xp[0]; xa1 = xp[1]; xa2 = xp[2]; xa3 = xp[3];
        const int4* pp = reinterpret_cast<const int4*>(prow + kt * 16);
        w0 = pp[0]; w1 = pp[1];
        float sc = srow[kt >> 2];                  // group = kt*32/128
        _Float16 sch  = (_Float16)sc;
        _Float16 offh = (_Float16)(sc * -1032.0f); // (1024+n)*sc - 1032*sc = (n-8)*sc
        sc2  = (h2){sch, sch};
        off2 = (h2){offh, offh};
    };

    auto lstore = [&](int buf) {
        // x tile: packed f32 -> f16 via v_cvt_pk_rtz_f16_f32
        V8u a0, a1;
        a0.h[0] = CVT_PK(xa0.x, xa0.y);
        a0.h[1] = CVT_PK(xa0.z, xa0.w);
        a0.h[2] = CVT_PK(xa1.x, xa1.y);
        a0.h[3] = CVT_PK(xa1.z, xa1.w);
        a1.h[0] = CVT_PK(xa2.x, xa2.y);
        a1.h[1] = CVT_PK(xa2.z, xa2.w);
        a1.h[2] = CVT_PK(xa3.x, xa3.y);
        a1.h[3] = CVT_PK(xa3.z, xa3.w);
        v8h* da = reinterpret_cast<v8h*>(&As[buf][r * LDT + ch * 16]);
        da[0] = a0.v; da[1] = a1.v;

        // W tile: int4 dequant via fp16 bit trick + v_pk_fma_f16
        // low nibble -> even col (low half), high nibble -> odd col (high half)
        const int bs[8] = {w0.x, w0.y, w0.z, w0.w, w1.x, w1.y, w1.z, w1.w};
        V8u b0, b1;
#pragma unroll
        for (int j = 0; j < 4; ++j) {
            unsigned b = (unsigned)bs[j];
            unsigned u = (b & 0xFu) | ((b & 0xF0u) << 12) | 0x64006400u;
            b0.h[j] = __builtin_bit_cast(h2, u) * sc2 + off2;
        }
#pragma unroll
        for (int j = 0; j < 4; ++j) {
            unsigned b = (unsigned)bs[4 + j];
            unsigned u = (b & 0xFu) | ((b & 0xF0u) << 12) | 0x64006400u;
            b1.h[j] = __builtin_bit_cast(h2, u) * sc2 + off2;
        }
        v8h* db = reinterpret_cast<v8h*>(&Bs[buf][r * LDT + ch * 16]);
        db[0] = b0.v; db[1] = b1.v;
    };

    auto compute = [&](int buf) {
        v16h af[2], bf[4];
#pragma unroll
        for (int tm = 0; tm < 2; ++tm) {
            const _Float16* base = &As[buf][(wm + tm * 16 + lrow) * LDT + lhalf * 8];
            v8h lo = *reinterpret_cast<const v8h*>(base);        // K 0-7 / 8-15
            v8h hi = *reinterpret_cast<const v8h*>(base + 16);   // K 16-23 / 24-31
            af[tm] = __builtin_shufflevector(lo, hi, 0, 1, 2, 3, 4, 5, 6, 7,
                                             8, 9, 10, 11, 12, 13, 14, 15);
        }
#pragma unroll
        for (int tn = 0; tn < 4; ++tn) {
            const _Float16* base = &Bs[buf][(wn + tn * 16 + lrow) * LDT + lhalf * 8];
            v8h lo = *reinterpret_cast<const v8h*>(base);
            v8h hi = *reinterpret_cast<const v8h*>(base + 16);
            bf[tn] = __builtin_shufflevector(lo, hi, 0, 1, 2, 3, 4, 5, 6, 7,
                                             8, 9, 10, 11, 12, 13, 14, 15);
        }
#pragma unroll
        for (int tm = 0; tm < 2; ++tm)
#pragma unroll
            for (int tn = 0; tn < 4; ++tn)
                acc[tm][tn] = __builtin_amdgcn_wmma_f32_16x16x32_f16(
                    false, af[tm], false, bf[tn],
                    (short)0, acc[tm][tn], false, false);
    };

    gload(0);
    lstore(0);

    for (int kt = 0; kt < NT; ++kt) {
        __syncthreads();
        if (kt + 1 < NT) gload(kt + 1);
        if (kt + 2 < NT) {
            __builtin_prefetch(xrow + (kt + 2) * BK, 0, 1);   // global_prefetch_b8
            __builtin_prefetch(prow + (kt + 2) * 16, 0, 1);
        }
        compute(kt & 1);
        if (kt + 1 < NT) lstore((kt + 1) & 1);
    }

    // C/D layout: VGPR i -> M = i + 8*(lane/16); N = lane%16
#pragma unroll
    for (int tm = 0; tm < 2; ++tm) {
#pragma unroll
        for (int tn = 0; tn < 4; ++tn) {
            const int row = bm + wm + tm * 16 + lhalf * 8;
            const int col = bn + wn + tn * 16 + lrow;
            float* op = out + (size_t)row * N_DIM + col;
#pragma unroll
            for (int i = 0; i < 8; ++i)
                op[(size_t)i * N_DIM] = acc[tm][tn][i];
        }
    }
}

extern "C" void kernel_launch(void* const* d_in, const int* in_sizes, int n_in,
                              void* d_out, int out_size, void* d_ws, size_t ws_size,
                              hipStream_t stream) {
    const float* x      = (const float*)d_in[0];
    const int*   packed = (const int*)d_in[1];
    const float* scales = (const float*)d_in[2];
    float*       out    = (float*)d_out;

    dim3 grid(N_DIM / TN_TILE, M_DIM / TM_TILE);  // 86 x 64
    dim3 block(256);
    QuantizedLinearZSE_59373627900306_kernel<<<grid, block, 0, stream>>>(x, packed, scales, out);
}
